// CoarseMatching_73607149519181
// MI455X (gfx1250) — matrix operand order repaired and checked by reading, not compiled, not run on hardware
//
#include <hip/hip_runtime.h>
#include <hip/hip_bf16.h>

// ---------------------------------------------------------------------------
// Problem constants (from reference setup_inputs)
// ---------------------------------------------------------------------------
#define NB   4
#define LL   4800      // h0c*w0c
#define SS   4800      // h1c*w1c
#define CC   256
#define H0   60
#define W0   80
#define THR  0.2f
#define BORD 2
#define SIM_SCALE  (1.0f / 25.6f)   // 1/(C*T) = 1/(256*0.1) applied to raw dot
#define KPT_SCALE  8.0f             // hw0_i_h / h0c = 480/60
#define NCHUNK 10
#define CHUNKL (LL / NCHUNK)        // 480

typedef __attribute__((ext_vector_type(16))) __bf16 v16bf;
typedef __attribute__((ext_vector_type(8)))  __bf16 v8bf;
typedef __attribute__((ext_vector_type(8)))  float  v8f;

// ---------------------------------------------------------------------------
// 1. f32 -> bf16 conversion (features fit in L2 afterwards: 9.8 MB each)
// ---------------------------------------------------------------------------
__global__ __launch_bounds__(256)
void k_cvt(const float* __restrict__ in, __hip_bfloat16* __restrict__ out, int n) {
    int i = blockIdx.x * blockDim.x + threadIdx.x;
    if (i < n) out[i] = __float2bfloat16(in[i]);
}

// ---------------------------------------------------------------------------
// 2. WMMA bf16 GEMM: sim[n] = f0[n] (LLxCC) * f1[n]^T (CCxSS), scaled.
//    Block = 64 threads = 2 waves stacked in M. Wave tile = 32 rows x 64 cols:
//    2 A-frags x 4 B-frags -> 8 WMMAs per k-step, software-pipelined so next
//    step's fragments are in flight while current WMMAs execute.
//    A/B fragment layouts per CDNA5 ISA 7.12.2 (16-bit A 16x32 / B 32x16).
// ---------------------------------------------------------------------------
__global__ __launch_bounds__(64)
void k_gemm(const __bf16* __restrict__ f0, const __bf16* __restrict__ f1,
            float* __restrict__ sim) {
    const int lane = threadIdx.x & 31;
    const int wave = threadIdx.x >> 5;           // 0..1
    const int n    = blockIdx.z;
    const int rowbase = blockIdx.y * 64 + wave * 32;
    const int colbase = blockIdx.x * 64;

    const __bf16* A = f0 + (size_t)n * LL * CC;
    const __bf16* B = f1 + (size_t)n * SS * CC;

    // A 16x32 bf16 layout: lanes 0-15 hold row=lane,    K {0..7} then {16..23};
    //                      lanes 16-31 hold row=lane-16, K {8..15} then {24..31}.
    const int halfa = (lane >> 4) * 8;
    // B 32x16 bf16 layout: lanes 0-15 col=lane, K {0..7} then {16..23}; lanes 16-31 K {8..15} then {24..31}.
    const int halfb = (lane >> 4) * 8;

    const __bf16* aptr = A + (size_t)(rowbase + (lane & 15)) * CC + halfa;
    const __bf16* bptr = B + (size_t)(colbase + (lane & 15)) * CC + halfb;

    v8f acc[2][4];
    #pragma unroll
    for (int i = 0; i < 2; ++i)
        #pragma unroll
        for (int j = 0; j < 4; ++j)
            acc[i][j] = (v8f){};

    auto load_frags = [&](v16bf (&af)[2], v16bf (&bf)[4], int k0) {
        #pragma unroll
        for (int i = 0; i < 2; ++i) {
            v8bf lo = *(const v8bf*)(aptr + (size_t)i * 16 * CC + k0);
            v8bf hi = *(const v8bf*)(aptr + (size_t)i * 16 * CC + k0 + 16);
            af[i] = __builtin_shufflevector(lo, hi,
                        0,1,2,3,4,5,6,7,8,9,10,11,12,13,14,15);
        }
        #pragma unroll
        for (int j = 0; j < 4; ++j)
            bf[j] = __builtin_shufflevector(
                *(const v8bf*)(bptr + (size_t)j * 16 * CC + k0),
                *(const v8bf*)(bptr + (size_t)j * 16 * CC + k0 + 16),
                0,1,2,3,4,5,6,7,8,9,10,11,12,13,14,15);
    };

    v16bf a_cur[2], b_cur[4], a_nxt[2], b_nxt[4];
    load_frags(a_cur, b_cur, 0);

    #pragma unroll
    for (int step = 0; step < CC / 32; ++step) {
        if (step < CC / 32 - 1)
            load_frags(a_nxt, b_nxt, (step + 1) * 32);   // overlap with WMMAs below

        #pragma unroll
        for (int i = 0; i < 2; ++i)
            #pragma unroll
            for (int j = 0; j < 4; ++j)
                acc[i][j] = __builtin_amdgcn_wmma_f32_16x16x32_bf16(
                    false, a_cur[i], false, b_cur[j],
                    (short)0, acc[i][j], false, false);

        #pragma unroll
        for (int i = 0; i < 2; ++i) a_cur[i] = a_nxt[i];
        #pragma unroll
        for (int j = 0; j < 4; ++j) b_cur[j] = b_nxt[j];
    }

    // C/D layout: VGPR r -> lanes 0-15: M=r, lanes 16-31: M=r+8; N=lane&15.
    float* out = sim + (size_t)n * LL * SS;
    const int c = colbase + (lane & 15);
    #pragma unroll
    for (int i = 0; i < 2; ++i) {
        const int r0 = rowbase + i * 16 + ((lane >> 4) * 8);
        #pragma unroll
        for (int r = 0; r < 8; ++r) {
            float* o = out + (size_t)(r0 + r) * SS + c;
            o[0]  = acc[i][0][r] * SIM_SCALE;
            o[16] = acc[i][1][r] * SIM_SCALE;
            o[32] = acc[i][2][r] * SIM_SCALE;
            o[48] = acc[i][3][r] * SIM_SCALE;
        }
    }
}

// ---------------------------------------------------------------------------
// 3. Row softmax stats (axis=2): one block per (n,l); online (m,d) + LDS merge
// ---------------------------------------------------------------------------
__global__ __launch_bounds__(256)
void k_row_stats(const float* __restrict__ sim,
                 float* __restrict__ rmax, float* __restrict__ rsuminv) {
    const int row = blockIdx.x;                    // n*LL + l
    const float* p = sim + (size_t)row * SS;
    float m = -1e30f, d = 0.f;
    for (int s = threadIdx.x; s < SS; s += 256) {
        float x = p[s];
        if (x > m) { d = d * __expf(m - x) + 1.f; m = x; }
        else       { d += __expf(x - m); }
    }
    __shared__ float sm[256], sd[256];
    sm[threadIdx.x] = m; sd[threadIdx.x] = d;
    __syncthreads();
    for (int off = 128; off > 0; off >>= 1) {
        if ((int)threadIdx.x < off) {
            float m2 = sm[threadIdx.x + off], d2 = sd[threadIdx.x + off];
            float M  = fmaxf(sm[threadIdx.x], m2);
            sd[threadIdx.x] = sd[threadIdx.x] * __expf(sm[threadIdx.x] - M)
                            + d2 * __expf(m2 - M);
            sm[threadIdx.x] = M;
        }
        __syncthreads();
    }
    if (threadIdx.x == 0) { rmax[row] = sm[0]; rsuminv[row] = 1.f / sd[0]; }
}

// ---------------------------------------------------------------------------
// 4. Column softmax stats (axis=1): thread-per-column (coalesced), L chunked
// ---------------------------------------------------------------------------
__global__ __launch_bounds__(256)
void k_col_partial(const float* __restrict__ sim,
                   float* __restrict__ pm, float* __restrict__ pd) {
    const int s = blockIdx.x * 256 + threadIdx.x;
    if (s >= SS) return;
    const int chunk = blockIdx.y, n = blockIdx.z;
    const float* base = sim + (size_t)n * LL * SS + (size_t)chunk * CHUNKL * SS + s;
    float m = -1e30f, d = 0.f;
    for (int l = 0; l < CHUNKL; ++l) {
        float x = base[(size_t)l * SS];
        if (x > m) { d = d * __expf(m - x) + 1.f; m = x; }
        else       { d += __expf(x - m); }
    }
    const size_t o = ((size_t)n * NCHUNK + chunk) * SS + s;
    pm[o] = m; pd[o] = d;
}

__global__ __launch_bounds__(256)
void k_col_combine(const float* __restrict__ pm, const float* __restrict__ pd,
                   float* __restrict__ cmax, float* __restrict__ csuminv) {
    const int i = blockIdx.x * 256 + threadIdx.x;   // n*SS + s
    if (i >= NB * SS) return;
    const int n = i / SS, s = i % SS;
    float m = -1e30f, d = 0.f;
    for (int c = 0; c < NCHUNK; ++c) {
        const size_t o = ((size_t)n * NCHUNK + c) * SS + s;
        float m2 = pm[o], d2 = pd[o];
        float M  = fmaxf(m, m2);
        d = d * __expf(m - M) + d2 * __expf(m2 - M);
        m = M;
    }
    cmax[i] = m; csuminv[i] = 1.f / d;
}

// ---------------------------------------------------------------------------
// 5. conf = exp(2*sim - rmax - cmax) * rsuminv * csuminv  (in-place over sim)
//    fused with per-row argmax
// ---------------------------------------------------------------------------
__global__ __launch_bounds__(256)
void k_conf(float* __restrict__ simconf,
            const float* __restrict__ rmax, const float* __restrict__ rsuminv,
            const float* __restrict__ cmax, const float* __restrict__ csuminv,
            float* __restrict__ rbv, int* __restrict__ rbj) {
    const int row = blockIdx.x;                    // n*LL + l
    const int n   = row / LL;
    const float rm = rmax[row], ri = rsuminv[row];
    float* p = simconf + (size_t)row * SS;
    const float* cm = cmax    + (size_t)n * SS;
    const float* ci = csuminv + (size_t)n * SS;
    float bv = -1.f; int bj = 0;
    for (int s = threadIdx.x; s < SS; s += 256) {
        float x = p[s];
        float c = __expf(2.f * x - rm - cm[s]) * ri * ci[s];
        p[s] = c;
        if (c > bv) { bv = c; bj = s; }
    }
    __shared__ float sv[256]; __shared__ int sj[256];
    sv[threadIdx.x] = bv; sj[threadIdx.x] = bj;
    __syncthreads();
    for (int off = 128; off > 0; off >>= 1) {
        if ((int)threadIdx.x < off && sv[threadIdx.x + off] > sv[threadIdx.x]) {
            sv[threadIdx.x] = sv[threadIdx.x + off];
            sj[threadIdx.x] = sj[threadIdx.x + off];
        }
        __syncthreads();
    }
    if (threadIdx.x == 0) { rbv[row] = sv[0]; rbj[row] = sj[0]; }
}

// ---------------------------------------------------------------------------
// 6. Column max of conf (mutual-nearest check): chunked + combine
// ---------------------------------------------------------------------------
__global__ __launch_bounds__(256)
void k_colbest_partial(const float* __restrict__ conf, float* __restrict__ pcm) {
    const int s = blockIdx.x * 256 + threadIdx.x;
    if (s >= SS) return;
    const int chunk = blockIdx.y, n = blockIdx.z;
    const float* base = conf + (size_t)n * LL * SS + (size_t)chunk * CHUNKL * SS + s;
    float m = -1.f;
    for (int l = 0; l < CHUNKL; ++l) m = fmaxf(m, base[(size_t)l * SS]);
    pcm[((size_t)n * NCHUNK + chunk) * SS + s] = m;
}

__global__ __launch_bounds__(256)
void k_colbest_combine(const float* __restrict__ pcm, float* __restrict__ cbv) {
    const int i = blockIdx.x * 256 + threadIdx.x;   // n*SS + s
    if (i >= NB * SS) return;
    const int n = i / SS, s = i % SS;
    float m = -1.f;
    for (int c = 0; c < NCHUNK; ++c)
        m = fmaxf(m, pcm[((size_t)n * NCHUNK + c) * SS + s]);
    cbv[i] = m;
}

// ---------------------------------------------------------------------------
// 7. Match extraction: threshold + border + mutual-nearest, keypoint coords
// ---------------------------------------------------------------------------
__device__ __forceinline__ bool border_ok(int idx) {
    const int r = idx / W0, c = idx % W0;
    return (r >= BORD) & (r < H0 - BORD) & (c >= BORD) & (c < W0 - BORD);
}

__global__ __launch_bounds__(256)
void k_final(const float* __restrict__ rbv, const int* __restrict__ rbj,
             const float* __restrict__ cbv,
             float* __restrict__ mask_v, float* __restrict__ all_j,
             float* __restrict__ mconf, float* __restrict__ mk1) {
    const int i = blockIdx.x * 256 + threadIdx.x;   // n*LL + l
    if (i >= NB * LL) return;
    const int n = i / LL, l = i % LL;
    const float v = rbv[i];
    const int   j = rbj[i];
    const bool ok = (v > THR) && border_ok(l) && border_ok(j)
                    && (v >= cbv[(size_t)n * SS + j]);
    const int jo = ok ? j : 0;
    mask_v[i] = ok ? 1.f : 0.f;
    all_j[i]  = (float)jo;
    mconf[i]  = ok ? v : 0.f;
    mk1[2 * i + 0] = (float)(jo % W0) * KPT_SCALE;
    mk1[2 * i + 1] = (float)(jo / W0) * KPT_SCALE;
}

__global__ __launch_bounds__(256)
void k_mk0(float* __restrict__ mk0) {
    const int l = blockIdx.x * 256 + threadIdx.x;
    if (l >= LL) return;
    mk0[2 * l + 0] = (float)(l % W0) * KPT_SCALE;
    mk0[2 * l + 1] = (float)(l / W0) * KPT_SCALE;
}

// ---------------------------------------------------------------------------
// Host launcher
// ---------------------------------------------------------------------------
extern "C" void kernel_launch(void* const* d_in, const int* in_sizes, int n_in,
                              void* d_out, int out_size, void* d_ws, size_t ws_size,
                              hipStream_t stream) {
    const float* f0 = (const float*)d_in[0];
    const float* f1 = (const float*)d_in[1];
    // d_in[2..6] are compile-time-known scalars (h0c,w0c,h1c,w1c,hw0_i_h) - baked in.

    float* out    = (float*)d_out;
    float* conf   = out;                                  // [NB,LL,SS] (sim in-place)
    float* mask_v = out + (size_t)NB * LL * SS;           // [NB,LL]
    float* all_j  = mask_v + (size_t)NB * LL;             // [NB,LL]
    float* mconf  = all_j  + (size_t)NB * LL;             // [NB,LL]
    float* mk0    = mconf  + (size_t)NB * LL;             // [LL,2]
    float* mk1    = mk0    + (size_t)LL * 2;              // [NB,LL,2]

    // Workspace layout
    char* ws = (char*)d_ws;
    __hip_bfloat16* f0b = (__hip_bfloat16*)ws;                          // NB*LL*CC bf16
    __hip_bfloat16* f1b = f0b + (size_t)NB * LL * CC;                   // NB*SS*CC bf16
    float* fw = (float*)(f1b + (size_t)NB * SS * CC);
    float* rmax    = fw; fw += (size_t)NB * LL;
    float* rsuminv = fw; fw += (size_t)NB * LL;
    float* cmax    = fw; fw += (size_t)NB * SS;
    float* csuminv = fw; fw += (size_t)NB * SS;
    float* rbv     = fw; fw += (size_t)NB * LL;
    int*   rbj     = (int*)fw; fw += (size_t)NB * LL;
    float* cbv     = fw; fw += (size_t)NB * SS;
    float* pm      = fw; fw += (size_t)NB * NCHUNK * SS;
    float* pd      = fw; fw += (size_t)NB * NCHUNK * SS;
    float* pcm     = fw; fw += (size_t)NB * NCHUNK * SS;

    const int nelem = NB * LL * CC;
    k_cvt<<<(nelem + 255) / 256, 256, 0, stream>>>(f0, f0b, nelem);
    k_cvt<<<(nelem + 255) / 256, 256, 0, stream>>>(f1, f1b, nelem);

    // GEMM: grid (col tiles 4800/64, row tiles 4800/64, batch), 2 waves/block
    k_gemm<<<dim3(SS / 64, LL / 64, NB), 64, 0, stream>>>(
        (const __bf16*)f0b, (const __bf16*)f1b, conf);

    k_row_stats<<<NB * LL, 256, 0, stream>>>(conf, rmax, rsuminv);

    k_col_partial<<<dim3((SS + 255) / 256, NCHUNK, NB), 256, 0, stream>>>(conf, pm, pd);
    k_col_combine<<<(NB * SS + 255) / 256, 256, 0, stream>>>(pm, pd, cmax, csuminv);

    k_conf<<<NB * LL, 256, 0, stream>>>(conf, rmax, rsuminv, cmax, csuminv, rbv, rbj);

    k_colbest_partial<<<dim3((SS + 255) / 256, NCHUNK, NB), 256, 0, stream>>>(conf, pcm);
    k_colbest_combine<<<(NB * SS + 255) / 256, 256, 0, stream>>>(pcm, cbv);

    k_final<<<(NB * LL + 255) / 256, 256, 0, stream>>>(rbv, rbj, cbv,
                                                       mask_v, all_j, mconf, mk1);
    k_mk0<<<(LL + 255) / 256, 256, 0, stream>>>(mk0);
}
